// TabrModelContextFreeze_71390946394126
// MI455X (gfx1250) — compile-verified, matrix operand order checked by reading, not run on hardware
//
// TabR-style context model for MI455X (gfx1250, CDNA5, wave32).
//
// Strategy (reasoned from MI455X specs, compile-only):
//  * ~0.7 TFLOP of GEMM math over ~1.5 GB streamed -> use bf16 WMMA
//    (v_wmma_f32_16x16x32_bf16, fp32 accum) for every GEMM; weights (~11 MB
//    bf16, pre-transposed [N][K]) stay resident in the 192 MB L2.
//  * GEMM: 128x256x32 block tile, 8 waves (2x4), each wave 64x64 (16 accums,
//    1 ds_load_b128 per WMMA). Double-buffered pipeline: the Tensor Data
//    Mover DMAs BOTH the A and B tiles of step i+1 into the alternate LDS
//    buffer (pad_enable -> padded [rows][40] layout, tensor_dim OOB
//    zero-fill at boundaries) while waves compute step i. One
//    s_wait_tensorcnt(0) + one barrier per K-step; zero VMEM in the hot loop.
//  * Workspace (~700 MB) bump-allocated from d_ws with region aliasing.

#include <hip/hip_runtime.h>
#include <hip/hip_bf16.h>
#include <stdint.h>

typedef __bf16 bf16_t;
typedef float  v8f   __attribute__((ext_vector_type(8)));
typedef bf16_t v16bf __attribute__((ext_vector_type(16)));
typedef bf16_t v8bf  __attribute__((ext_vector_type(8)));
typedef unsigned int u32;
typedef u32 u32x4 __attribute__((ext_vector_type(4)));
typedef int i32x4 __attribute__((ext_vector_type(4)));
typedef int i32x8 __attribute__((ext_vector_type(8)));

#if defined(__has_builtin)
#if __has_builtin(__builtin_amdgcn_tensor_load_to_lds) && __has_builtin(__builtin_amdgcn_s_wait_tensorcnt)
#define HAVE_TDM 1
#endif
#endif
#ifndef HAVE_TDM
#define HAVE_TDM 0
#endif

#define BM 128
#define BN 256
#define BK 32
#define LDSP 40  // 32 + 8 bf16 pad (16B): matches TDM pad interval 16dw / amount 4dw

#define NEG_HUGE (-3.402823466e38f)

// ---------------------------------------------------------------------------
// TDM: DMA a [tileRows] x [32] bf16 tile (row stride K elems) into LDS,
// padded to LDSP row stride. Rows beyond tensorRows are OOB -> zero-filled.
// 2-D descriptor per CDNA5 ISA.
// ---------------------------------------------------------------------------
#if HAVE_TDM
__device__ __forceinline__ void tdm_load_tile_bf16(const bf16_t* gptr, u32 ldsByteOff,
                                                   u32 K, u32 tensorRows, u32 tileRows) {
  unsigned long long ga = (unsigned long long)(uintptr_t)gptr;
  u32x4 g0;
  g0[0] = 1u;                                   // count=1 (valid descriptor)
  g0[1] = ldsByteOff;                           // lds_addr
  g0[2] = (u32)(ga & 0xffffffffu);              // global_addr[31:0]
  g0[3] = (u32)((ga >> 32) & 0x1ffffffu) | (2u << 30);  // addr[56:32], type=2
  // group1: data_size=2B(code 1), pad_enable, pad_interval=16dw(code 3),
  // pad_amount=4dw(code 3); tensor_dim0=K, tensor_dim1=tensorRows,
  // tile_dim0=32, tile_dim1=tileRows, tensor_dim0_stride=K.
  u32 w0 = (1u << 16) | (1u << 20) | (3u << 22) | (3u << 25);
  u32 w1 = (K & 0xffffu) << 16;
  u32 w2 = ((K >> 16) & 0xffffu) | ((tensorRows & 0xffffu) << 16);
  u32 w3 = ((tensorRows >> 16) & 0xffffu) | (32u << 16);
  u32 w4 = tileRows & 0xffffu;
  u32 w5 = K;
  i32x8 g1;
  g1[0] = (int)w0; g1[1] = (int)w1; g1[2] = (int)w2; g1[3] = (int)w3;
  g1[4] = (int)w4; g1[5] = (int)w5; g1[6] = 0; g1[7] = 0;
  i32x4 gz = {};
#if __clang_major__ >= 23
  i32x8 gz8 = {};
  __builtin_amdgcn_tensor_load_to_lds(g0, g1, gz, gz, gz8, 0);
#else
  __builtin_amdgcn_tensor_load_to_lds(g0, g1, gz, gz, 0);
#endif
}
#endif

// ---------------------------------------------------------------------------
// Generic bf16 WMMA GEMM: out = epilogue( scale * A[M,K] @ Bt[N,K]^T )
//   epilogue: + bias[col] (opt) + resid[row,col] (opt), ReLU (opt),
//   writes fp32 outF and/or bf16 outH. K must be a multiple of 32.
// ---------------------------------------------------------------------------
__global__ __launch_bounds__(256) void gemm_bf16_kernel(
    const bf16_t* __restrict__ A, const bf16_t* __restrict__ Bt,
    const float* __restrict__ bias, const float* __restrict__ resid,
    float* __restrict__ outF, bf16_t* __restrict__ outH,
    int M, int N, int K, float scale, int relu) {
  __shared__ bf16_t As[2][BM][LDSP];
  __shared__ bf16_t Bs[2][BN][LDSP];

  const int tid = threadIdx.x;
  const int wave = tid >> 5, lane = tid & 31;
  const int lr = lane & 15, hsel = lane >> 4;
  const int waveM = wave >> 2, waveN = wave & 3;  // 2 x 4 wave grid, 64x64/wave
  const int m0 = blockIdx.y * BM, n0 = blockIdx.x * BN;
  const int nk = K / BK;

  v8f acc[4][4] = {};
  union Frag { v16bf v; v8bf h[2]; };

  // CDNA5 wave32 16-bit fragment layout:
  // lane<16: row=lane, K {0..7,16..23}; lane>=16: row=lane-16, K {8..15,24..31}.
  auto compute = [&](int cur) {
    Frag af[4], bfg[4];
#pragma unroll
    for (int mi = 0; mi < 4; mi++) {
      const bf16_t* p = &As[cur][waveM * 64 + mi * 16 + lr][hsel * 8];
      af[mi].h[0] = *(const v8bf*)p;
      af[mi].h[1] = *(const v8bf*)(p + 16);
    }
#pragma unroll
    for (int ni = 0; ni < 4; ni++) {
      const bf16_t* p = &Bs[cur][waveN * 64 + ni * 16 + lr][hsel * 8];
      bfg[ni].h[0] = *(const v8bf*)p;
      bfg[ni].h[1] = *(const v8bf*)(p + 16);
    }
#pragma unroll
    for (int mi = 0; mi < 4; mi++)
#pragma unroll
      for (int ni = 0; ni < 4; ni++)
        acc[mi][ni] = __builtin_amdgcn_wmma_f32_16x16x32_bf16(
            false, af[mi].v, false, bfg[ni].v, (short)0, acc[mi][ni], false, false);
  };

#if HAVE_TDM
  // TDM-fed double-buffered pipeline: DMA tile i+1 while computing tile i.
  auto issue = [&](int kt, int buf) {
    if (wave == 0) {
      int k0 = kt * BK;
      tdm_load_tile_bf16(A + (size_t)m0 * K + k0,
                         (u32)(uintptr_t)&As[buf][0][0], (u32)K,
                         (u32)(M - m0), (u32)BM);
      tdm_load_tile_bf16(Bt + (size_t)n0 * K + k0,
                         (u32)(uintptr_t)&Bs[buf][0][0], (u32)K,
                         (u32)(N - n0), (u32)BN);
    }
  };
  issue(0, 0);
  __builtin_amdgcn_s_wait_tensorcnt(0);
  __syncthreads();
  for (int i = 0; i < nk; i++) {
    int cur = i & 1;
    if (i + 1 < nk) issue(i + 1, cur ^ 1);   // overlap DMA with compute
    compute(cur);
    __builtin_amdgcn_s_wait_tensorcnt(0);
    __syncthreads();
  }
#else
  // Fallback: register-double-buffered cooperative loads.
  v8bf zero = {};
  v8bf va[2], vb[4];
  auto load_regs = [&](int kt) {
    int k0 = kt * BK;
#pragma unroll
    for (int j = 0; j < 2; j++) {
      int c = tid + j * 256, r = c >> 2, cc = (c & 3) * 8;
      int gr = m0 + r;
      va[j] = zero;
      if (gr < M) va[j] = *(const v8bf*)(A + (size_t)gr * K + k0 + cc);
    }
#pragma unroll
    for (int j = 0; j < 4; j++) {
      int c = tid + j * 256, r = c >> 2, cc = (c & 3) * 8;
      int gn = n0 + r;
      vb[j] = zero;
      if (gn < N) vb[j] = *(const v8bf*)(Bt + (size_t)gn * K + k0 + cc);
    }
  };
  auto store_regs = [&](int buf) {
#pragma unroll
    for (int j = 0; j < 2; j++) {
      int c = tid + j * 256, r = c >> 2, cc = (c & 3) * 8;
      *(v8bf*)(&As[buf][r][cc]) = va[j];
    }
#pragma unroll
    for (int j = 0; j < 4; j++) {
      int c = tid + j * 256, r = c >> 2, cc = (c & 3) * 8;
      *(v8bf*)(&Bs[buf][r][cc]) = vb[j];
    }
  };
  load_regs(0);
  store_regs(0);
  __syncthreads();
  for (int i = 0; i < nk; i++) {
    int cur = i & 1;
    if (i + 1 < nk) load_regs(i + 1);
    compute(cur);
    __syncthreads();
    if (i + 1 < nk) store_regs(cur ^ 1);
    __syncthreads();
  }
#endif

  // Epilogue: C/D layout — VGPR i: M = i + (lane>=16 ? 8 : 0), N = lane&15.
#pragma unroll
  for (int mi = 0; mi < 4; mi++) {
#pragma unroll
    for (int ni = 0; ni < 4; ni++) {
      int col = n0 + waveN * 64 + ni * 16 + lr;
      int rowb = m0 + waveM * 64 + mi * 16 + hsel * 8;
#pragma unroll
      for (int i = 0; i < 8; i++) {
        int row = rowb + i;
        if (row < M && col < N) {
          size_t o = (size_t)row * N + col;
          float v = acc[mi][ni][i] * scale;
          if (bias) v += bias[col];
          if (resid) v += resid[o];
          if (relu) v = fmaxf(v, 0.0f);
          if (outF) outF[o] = v;
          if (outH) outH[o] = (bf16_t)v;
        }
      }
    }
  }
}

// ---------------------------------------------------------------------------
// LayerNorm over D=512, one block (128 threads) per row.
// ---------------------------------------------------------------------------
__global__ __launch_bounds__(128) void ln_kernel(
    const float* __restrict__ in, const float* __restrict__ g,
    const float* __restrict__ b, float* __restrict__ outF,
    bf16_t* __restrict__ outH, int M) {
  int row = blockIdx.x;
  if (row >= M) return;
  const float* x = in + (size_t)row * 512;
  int tid = threadIdx.x;
  float s = 0.f, s2 = 0.f;
  for (int d = tid; d < 512; d += 128) { float v = x[d]; s += v; s2 += v * v; }
  __shared__ float r1[128], r2[128];
  r1[tid] = s; r2[tid] = s2; __syncthreads();
  for (int st = 64; st > 0; st >>= 1) {
    if (tid < st) { r1[tid] += r1[tid + st]; r2[tid] += r2[tid + st]; }
    __syncthreads();
  }
  float m = r1[0] * (1.f / 512.f);
  float var = r2[0] * (1.f / 512.f) - m * m;
  float inv = rsqrtf(var + 1e-5f);
  for (int d = tid; d < 512; d += 128) {
    float v = (x[d] - m) * inv * g[d] + b[d];
    if (outF) outF[(size_t)row * 512 + d] = v;
    if (outH) outH[(size_t)row * 512 + d] = (bf16_t)v;
  }
}

// -||row||^2 over D=512 (bias for the scores GEMM)
__global__ __launch_bounds__(128) void negsq_kernel(const float* __restrict__ in,
                                                    float* __restrict__ out) {
  int row = blockIdx.x, tid = threadIdx.x;
  const float* x = in + (size_t)row * 512;
  float s = 0.f;
  for (int d = tid; d < 512; d += 128) { float v = x[d]; s += v * v; }
  __shared__ float r1[128];
  r1[tid] = s; __syncthreads();
  for (int st = 64; st > 0; st >>= 1) {
    if (tid < st) r1[tid] += r1[tid + st];
    __syncthreads();
  }
  if (tid == 0) out[row] = -r1[0];
}

__global__ void cvt_kernel(const float* __restrict__ in, bf16_t* __restrict__ out, int n) {
  int i = blockIdx.x * blockDim.x + threadIdx.x;
  if (i < n) out[i] = (bf16_t)in[i];
}

// weight transpose+convert: in[K,N] fp32 -> out[N,K] bf16
__global__ void wtr_kernel(const float* __restrict__ in, bf16_t* __restrict__ out,
                           int K, int N) {
  int i = blockIdx.x * blockDim.x + threadIdx.x;
  if (i < K * N) {
    int k = i / N, n = i % N;
    out[(size_t)n * K + k] = (bf16_t)in[i];
  }
}

// per-row iterative top-C argmax (destroys scores row); also emits abs idx
__global__ __launch_bounds__(256) void topk_kernel(
    float* __restrict__ scores, const int* __restrict__ cand_idx,
    int* __restrict__ ctx_idx, int* __restrict__ abs_idx, int N, int C) {
  int b = blockIdx.x, tid = threadIdx.x;
  float* row = scores + (size_t)b * N;
  __shared__ float sval[256];
  __shared__ int sidx[256];
  for (int it = 0; it < C; it++) {
    float best = NEG_HUGE; int bi = 0;
    for (int j = tid; j < N; j += 256) {
      float v = row[j];
      if (v > best) { best = v; bi = j; }
    }
    sval[tid] = best; sidx[tid] = bi; __syncthreads();
    for (int st = 128; st > 0; st >>= 1) {
      if (tid < st && sval[tid + st] > sval[tid]) {
        sval[tid] = sval[tid + st]; sidx[tid] = sidx[tid + st];
      }
      __syncthreads();
    }
    if (tid == 0) {
      int w = sidx[0];
      ctx_idx[b * C + it] = w;
      abs_idx[b * C + it] = cand_idx[w];
      row[w] = NEG_HUGE;
    }
    __syncthreads();
  }
}

// one wave per (b,c): diff = k[b]-ck[idx] (bf16 out), sims = -||diff||^2
__global__ __launch_bounds__(256) void diff_kernel(
    const float* __restrict__ qk, const float* __restrict__ ck,
    const int* __restrict__ ctx_idx, bf16_t* __restrict__ diffH,
    float* __restrict__ sims, int npairs) {
  int pair = blockIdx.x * 8 + (threadIdx.x >> 5);
  if (pair >= npairs) return;
  int lane = threadIdx.x & 31;
  int b = pair / 96;
  int idx = ctx_idx[pair];
  const float* kq = qk + (size_t)b * 512;
  const float* kc = ck + (size_t)idx * 512;
  float s = 0.f;
  for (int d = lane; d < 512; d += 32) {
    float df = kq[d] - kc[d];
    diffH[(size_t)pair * 512 + d] = (bf16_t)df;
    s += df * df;
  }
  for (int o = 16; o > 0; o >>= 1) s += __shfl_down(s, o);
  if (lane == 0) sims[pair] = -s;
}

__global__ __launch_bounds__(128) void softmax_kernel(
    const float* __restrict__ sims, float* __restrict__ probs,
    float* __restrict__ out_probs) {
  int b = blockIdx.x, tid = threadIdx.x;
  __shared__ float sv[96];
  __shared__ float red[2];
  if (tid < 96) sv[tid] = sims[b * 96 + tid];
  __syncthreads();
  if (tid == 0) {
    float m = NEG_HUGE;
    for (int c = 0; c < 96; c++) m = fmaxf(m, sv[c]);
    red[0] = m;
  }
  __syncthreads();
  if (tid < 96) sv[tid] = __expf(sv[tid] - red[0]);
  __syncthreads();
  if (tid == 0) {
    float s = 0.f;
    for (int c = 0; c < 96; c++) s += sv[c];
    red[1] = 1.0f / s;
  }
  __syncthreads();
  if (tid < 96) {
    float p = sv[tid] * red[1];
    probs[b * 96 + tid] = p;
    out_probs[b * 96 + tid] = p;
  }
}

// h0[b,d] = x[b,d] + sum_c p[c]*t[b,c,d] + (sum_c p[c]*y[idx])*w_label[d] + b_label[d]
__global__ __launch_bounds__(512) void combine_kernel(
    const float* __restrict__ probs, const int* __restrict__ ctx_idx,
    const float* __restrict__ cand_y, const float* __restrict__ w_label,
    const float* __restrict__ b_label, const float* __restrict__ tF,
    const float* __restrict__ xF, float* __restrict__ h0) {
  int b = blockIdx.x, tid = threadIdx.x;
  __shared__ float p[96], py[96];
  __shared__ float pys;
  if (tid < 96) {
    float pv = probs[b * 96 + tid];
    p[tid] = pv;
    py[tid] = pv * cand_y[ctx_idx[b * 96 + tid]];
  }
  __syncthreads();
  if (tid == 0) {
    float s = 0.f;
    for (int c = 0; c < 96; c++) s += py[c];
    pys = s;
  }
  __syncthreads();
  int d = tid;  // blockDim == 512 == D
  float acc = pys * w_label[d] + b_label[d];
  const float* tb = tF + ((size_t)b * 96) * 512 + d;
  for (int c = 0; c < 96; c++) acc += p[c] * tb[(size_t)c * 512];
  h0[(size_t)b * 512 + d] = xF[(size_t)b * 512 + d] + acc;
}

// y[b] = relu(LN(h[b])) . W_head + b_head
__global__ __launch_bounds__(128) void head_kernel(
    const float* __restrict__ h, const float* __restrict__ g,
    const float* __restrict__ bt, const float* __restrict__ Wh,
    const float* __restrict__ bh, float* __restrict__ y) {
  int row = blockIdx.x, tid = threadIdx.x;
  const float* x = h + (size_t)row * 512;
  float s = 0.f, s2 = 0.f;
  for (int d = tid; d < 512; d += 128) { float v = x[d]; s += v; s2 += v * v; }
  __shared__ float r1[128], r2[128];
  r1[tid] = s; r2[tid] = s2; __syncthreads();
  for (int st = 64; st > 0; st >>= 1) {
    if (tid < st) { r1[tid] += r1[tid + st]; r2[tid] += r2[tid + st]; }
    __syncthreads();
  }
  float m = r1[0] * (1.f / 512.f);
  float var = r2[0] * (1.f / 512.f) - m * m;
  float inv = rsqrtf(var + 1e-5f);
  __syncthreads();
  float acc = 0.f;
  for (int d = tid; d < 512; d += 128) {
    float v = (x[d] - m) * inv * g[d] + bt[d];
    acc += fmaxf(v, 0.f) * Wh[d];
  }
  r1[tid] = acc; __syncthreads();
  for (int st = 64; st > 0; st >>= 1) {
    if (tid < st) r1[tid] += r1[tid + st];
    __syncthreads();
  }
  if (tid == 0) y[row] = r1[0] + bh[0];
}

// ---------------------------------------------------------------------------
extern "C" void kernel_launch(void* const* d_in, const int* in_sizes, int n_in,
                              void* d_out, int out_size, void* d_ws, size_t ws_size,
                              hipStream_t stream) {
  (void)in_sizes; (void)n_in; (void)out_size; (void)ws_size;
  enum { Bq = 1024, Nc = 50000, Cc = 96, Din = 96, Dm = 512, Db = 1024 };

  const float* x_num   = (const float*)d_in[0];
  const float* cand_x  = (const float*)d_in[1];
  const float* cand_y  = (const float*)d_in[2];
  const int*   candIdx = (const int*)d_in[3];
  // d_in[4] = context_size (== 96, fixed at compile time)
  const float* W_in = (const float*)d_in[5];
  const float* b_in = (const float*)d_in[6];
  const float* encg = (const float*)d_in[7];
  const float* encb = (const float*)d_in[8];
  const float* eW1  = (const float*)d_in[9];
  const float* eb1  = (const float*)d_in[10];
  const float* eW2  = (const float*)d_in[11];
  const float* eb2  = (const float*)d_in[12];
  const float* mixg = (const float*)d_in[13];
  const float* mixb = (const float*)d_in[14];
  const float* Wk   = (const float*)d_in[15];
  const float* bk   = (const float*)d_in[16];
  const float* w_label = (const float*)d_in[17];
  const float* b_label = (const float*)d_in[18];
  const float* Wt1  = (const float*)d_in[19];
  const float* bt1  = (const float*)d_in[20];
  const float* Wt2  = (const float*)d_in[21];
  const float* pg   = (const float*)d_in[22];
  const float* pb   = (const float*)d_in[23];
  const float* pW1  = (const float*)d_in[24];
  const float* pb1  = (const float*)d_in[25];
  const float* pW2  = (const float*)d_in[26];
  const float* pb2  = (const float*)d_in[27];
  const float* hg   = (const float*)d_in[28];
  const float* hb   = (const float*)d_in[29];
  const float* Whead = (const float*)d_in[30];
  const float* bhead = (const float*)d_in[31];

  // ---- workspace bump allocator (256B aligned) -------------------------
  char* base = (char*)d_ws;
  size_t off = 0;
  auto alloc = [&](size_t bytes) -> char* {
    char* p = base + off;
    off = (off + bytes + 255) & ~(size_t)255;
    return p;
  };

  // bf16 transposed weights (persistent, ~11 MB -> lives in 192 MB L2)
  bf16_t* WinT = (bf16_t*)alloc((size_t)Din * Dm * 2);
  bf16_t *W1T[2], *W2T[2], *pW1T[2], *pW2T[2];
  for (int i = 0; i < 2; i++) W1T[i] = (bf16_t*)alloc((size_t)Dm * Db * 2);
  for (int i = 0; i < 2; i++) W2T[i] = (bf16_t*)alloc((size_t)Db * Dm * 2);
  bf16_t* WkT  = (bf16_t*)alloc((size_t)Dm * Dm * 2);
  bf16_t* Wt1T = (bf16_t*)alloc((size_t)Dm * Db * 2);
  bf16_t* Wt2T = (bf16_t*)alloc((size_t)Db * Dm * 2);
  for (int i = 0; i < 2; i++) pW1T[i] = (bf16_t*)alloc((size_t)Dm * Db * 2);
  for (int i = 0; i < 2; i++) pW2T[i] = (bf16_t*)alloc((size_t)Db * Dm * 2);

  bf16_t* xH = (bf16_t*)alloc((size_t)Bq * Din * 2);
  bf16_t* cH = (bf16_t*)alloc((size_t)Nc * Din * 2);

  float*  candKF = (float*)alloc((size_t)Nc * Dm * 4);
  bf16_t* candKH = (bf16_t*)alloc((size_t)Nc * Dm * 2);
  float*  negcsq = (float*)alloc((size_t)Nc * 4);
  float*  qKF = (float*)alloc((size_t)Bq * Dm * 4);
  bf16_t* qKH = (bf16_t*)alloc((size_t)Bq * Dm * 2);
  float*  xF  = (float*)alloc((size_t)Bq * Dm * 4);
  int*    ctxIdx = (int*)alloc((size_t)Bq * Cc * 4);
  float*  sims   = (float*)alloc((size_t)Bq * Cc * 4);
  float*  probs  = (float*)alloc((size_t)Bq * Cc * 4);
  float*  h0  = (float*)alloc((size_t)Bq * Dm * 4);
  bf16_t* lnP = (bf16_t*)alloc((size_t)Bq * Dm * 2);
  bf16_t* z1P = (bf16_t*)alloc((size_t)Bq * Db * 2);

  // scores region (204.8 MB) -> later reused as t [B*C,512] fp32 (201.3 MB)
  char* scoresR = alloc((size_t)Bq * Nc * 4);
  float* scores = (float*)scoresR;
  float* tF = (float*)scoresR;
  // encoder temp region (307.2 MB) -> later reused as diffH + z1H
  size_t rt = (size_t)Nc * Dm * 4 + (size_t)Nc * Dm * 2 + (size_t)Nc * Db * 2 +
              (size_t)Nc * Dm * 2;
  char* regionT = alloc(rt);
  float*  ehF  = (float*)regionT;
  bf16_t* ehH  = (bf16_t*)(regionT + (size_t)Nc * Dm * 4);
  bf16_t* et1H = (bf16_t*)(regionT + (size_t)Nc * Dm * 4 + (size_t)Nc * Dm * 2);
  bf16_t* elnH = (bf16_t*)(regionT + (size_t)Nc * Dm * 4 + (size_t)Nc * Dm * 2 +
                           (size_t)Nc * Db * 2);
  bf16_t* diffH = (bf16_t*)regionT;                                   // 100.7 MB
  bf16_t* z1H = (bf16_t*)(regionT + (size_t)Bq * Cc * Dm * 2);        // 201.3 MB

  // ---- launch helpers ---------------------------------------------------
  auto gemm = [&](const bf16_t* A, const bf16_t* Bt, const float* bias,
                  const float* resid, float* oF, bf16_t* oH,
                  int M, int N, int K, float scale, int relu) {
    dim3 g((N + BN - 1) / BN, (M + BM - 1) / BM);
    gemm_bf16_kernel<<<g, dim3(256), 0, stream>>>(A, Bt, bias, resid, oF, oH,
                                                  M, N, K, scale, relu);
  };
  auto ln = [&](const float* in, const float* g_, const float* b_, float* oF,
                bf16_t* oH, int M) {
    ln_kernel<<<M, 128, 0, stream>>>(in, g_, b_, oF, oH, M);
  };
  auto wtr = [&](const float* in, bf16_t* out, int K, int N) {
    wtr_kernel<<<(K * N + 255) / 256, 256, 0, stream>>>(in, out, K, N);
  };

  // ---- 1) convert + transpose weights, convert inputs -------------------
  wtr(W_in, WinT, Din, Dm);
  wtr(eW1, W1T[0], Dm, Db); wtr(eW1 + (size_t)Dm * Db, W1T[1], Dm, Db);
  wtr(eW2, W2T[0], Db, Dm); wtr(eW2 + (size_t)Db * Dm, W2T[1], Db, Dm);
  wtr(Wk, WkT, Dm, Dm);
  wtr(Wt1, Wt1T, Dm, Db);
  wtr(Wt2, Wt2T, Db, Dm);
  wtr(pW1, pW1T[0], Dm, Db); wtr(pW1 + (size_t)Dm * Db, pW1T[1], Dm, Db);
  wtr(pW2, pW2T[0], Db, Dm); wtr(pW2 + (size_t)Db * Dm, pW2T[1], Db, Dm);
  cvt_kernel<<<(Bq * Din + 255) / 256, 256, 0, stream>>>(x_num, xH, Bq * Din);
  cvt_kernel<<<(Nc * Din + 255) / 256, 256, 0, stream>>>(cand_x, cH, Nc * Din);

  // ---- 2) encoder (shared by candidates and queries) ---------------------
  auto encode = [&](const bf16_t* inH, int M, float* hF, bf16_t* hH, bf16_t* t1H,
                    bf16_t* lnH, float* kF, bf16_t* kH) {
    gemm(inH, WinT, b_in, nullptr, hF, hH, M, Dm, Din, 1.f, 0);   // h = x@W_in+b
    gemm(hH, W1T[0], eb1, nullptr, nullptr, t1H, M, Db, Dm, 1.f, 1);   // blk0 (no LN)
    gemm(t1H, W2T[0], eb2, hF, hF, nullptr, M, Dm, Db, 1.f, 0);        // h += z
    ln(hF, encg + Dm, encb + Dm, nullptr, lnH, M);                     // blk1 prenorm
    gemm(lnH, W1T[1], eb1 + Db, nullptr, nullptr, t1H, M, Db, Dm, 1.f, 1);
    gemm(t1H, W2T[1], eb2 + Dm, hF, hF, nullptr, M, Dm, Db, 1.f, 0);
    ln(hF, mixg, mixb, nullptr, lnH, M);                               // mix LN
    gemm(lnH, WkT, bk, nullptr, kF, kH, M, Dm, Dm, 1.f, 0);            // k
  };
  encode(cH, Nc, ehF, ehH, et1H, elnH, candKF, candKH);
  negsq_kernel<<<Nc, 128, 0, stream>>>(candKF, negcsq);
  encode(xH, Bq, xF /*keep x=h*/, ehH, et1H, elnH, qKF, qKH);

  // ---- 3) kNN: scores = 2*k@ck^T - ||ck||^2  (-||k||^2 is rank-invariant)
  gemm(qKH, candKH, negcsq, nullptr, scores, nullptr, Bq, Nc, Dm, 2.0f, 0);
  float* outY = (float*)d_out;
  int* outIdx = (int*)((float*)d_out + Bq);
  float* outProbs = (float*)d_out + Bq + (size_t)Bq * Cc;
  topk_kernel<<<Bq, 256, 0, stream>>>(scores, candIdx, ctxIdx, outIdx, Nc, Cc);

  // ---- 4) context: diff/sims -> softmax -> T-network -> combine ----------
  diff_kernel<<<(Bq * Cc + 7) / 8, 256, 0, stream>>>(qKF, candKF, ctxIdx, diffH,
                                                     sims, Bq * Cc);
  softmax_kernel<<<Bq, 128, 0, stream>>>(sims, probs, outProbs);
  gemm(diffH, Wt1T, bt1, nullptr, nullptr, z1H, Bq * Cc, Db, Dm, 1.f, 1);
  gemm(z1H, Wt2T, nullptr, nullptr, tF, nullptr, Bq * Cc, Dm, Db, 1.f, 0);
  combine_kernel<<<Bq, 512, 0, stream>>>(probs, ctxIdx, cand_y, w_label, b_label,
                                         tF, xF, h0);

  // ---- 5) predictor blocks + head ----------------------------------------
  for (int i = 0; i < 2; i++) {
    ln(h0, pg + i * Dm, pb + i * Dm, nullptr, lnP, Bq);
    gemm(lnP, pW1T[i], pb1 + i * Db, nullptr, nullptr, z1P, Bq, Db, Dm, 1.f, 1);
    gemm(z1P, pW2T[i], pb2 + i * Dm, h0, h0, nullptr, Bq, Dm, Db, 1.f, 0);
  }
  head_kernel<<<Bq, 128, 0, stream>>>(h0, hg, hb, Whead, bhead, outY);
}
